// SparseAttentionWrapper_82205674045541
// MI455X (gfx1250) — compile-verified, matrix operand order checked by reading, not run on hardware
//
#include <hip/hip_runtime.h>

typedef unsigned short u16;
typedef __attribute__((ext_vector_type(8)))  unsigned short u16x8;
typedef __attribute__((ext_vector_type(16))) __bf16        v16bf;
typedef __attribute__((ext_vector_type(8)))  float         v8f;
typedef __attribute__((ext_vector_type(4)))  unsigned int  u32x4;
typedef __attribute__((ext_vector_type(8)))  int           i32x8;
typedef __attribute__((ext_vector_type(4)))  int           i32x4;

union BF16Frag { v16bf v; u16 u[16]; };

// ---- constants (from reference) ----
#define BB   16
#define NN   1024
#define CC   768
#define HH   12
#define DD   64
#define MM   (BB * NN)      // 16384
#define K3   (3 * CC)       // 2304
#define STRIDE_L 14
#define CHUNK  128          // K elements staged per TDM transfer (GEMM)
#define NCHUNK (CC / CHUNK) // 6
#define NSLOT  192          // gathered key slots per query tile (6 chunks of 32)
#define NKC    (NSLOT / 32) // 6
#define SENT   0xFFFFu

#if __has_builtin(__builtin_amdgcn_tensor_load_to_lds)
#define HAVE_TDM 1
#else
#define HAVE_TDM 0
#endif

// ---------- bf16 <-> f32 ----------
__device__ inline u16 f2bf(float f) {
  union { float f; unsigned u; } c; c.f = f;
  unsigned u = c.u;
  return (u16)((u + 0x7FFFu + ((u >> 16) & 1u)) >> 16);  // RNE
}

// ---------- WMMA fragment load ----------
// 16-bit A/B layout (ISA 7.12.2): lane L holds row/col (L%16); half = L/16.
// => two contiguous 16-byte chunks at element offsets half*8 and 16+half*8.
__device__ inline v16bf load_frag(const u16* p, int hf) {
  u16x8 lo = *(const u16x8*)(p + hf * 8);
  u16x8 hi = *(const u16x8*)(p + 16 + hf * 8);
  BF16Frag r;
  #pragma unroll
  for (int i = 0; i < 8; ++i) { r.u[i] = lo[i]; r.u[i + 8] = hi[i]; }
  return r.v;
}

__device__ inline v8f wmma_bf16(v16bf a, v16bf b, v8f c) {
  return __builtin_amdgcn_wmma_f32_16x16x32_bf16(false, a, false, b, (short)0, c, false, false);
}

#if HAVE_TDM
// ---------- TDM: 2D tile load (rows x k-elems of bf16), row-major into LDS ----------
__device__ inline void tdm_load_2d(unsigned lds_off, const u16* gptr,
                                   unsigned tensor_d0, unsigned tensor_d1,
                                   unsigned tile_d0, unsigned tile_d1,
                                   unsigned stride_elems) {
  unsigned long long ga = (unsigned long long)(size_t)gptr;
  u32x4 g0;
  g0.x = 1u;                                   // count=1
  g0.y = lds_off;                              // lds_addr (bytes)
  g0.z = (unsigned)ga;                         // global_addr[31:0]
  g0.w = (unsigned)(ga >> 32) | (2u << 30);    // global_addr[56:32] | type=2
  i32x8 g1;
  g1[0] = (int)(1u << 16);                                          // data_size=1 (2B)
  g1[1] = (int)((tensor_d0 & 0xFFFFu) << 16);                       // tensor_dim0 lo
  g1[2] = (int)((tensor_d0 >> 16) | ((tensor_d1 & 0xFFFFu) << 16)); // td0 hi | td1 lo
  g1[3] = (int)((tensor_d1 >> 16) | (tile_d0 << 16));               // td1 hi | tile_dim0
  g1[4] = (int)(tile_d1 & 0xFFFFu);                                 // tile_dim1
  g1[5] = (int)stride_elems;                                        // dim0_stride lo
  g1[6] = 0;
  g1[7] = 0;
  i32x4 z4 = {0, 0, 0, 0};
#if defined(__clang_major__) && (__clang_major__ >= 23)
  i32x8 z8 = {0, 0, 0, 0, 0, 0, 0, 0};
  __builtin_amdgcn_tensor_load_to_lds(g0, g1, z4, z4, z8, 0);
#else
  __builtin_amdgcn_tensor_load_to_lds(g0, g1, z4, z4, 0);
#endif
}
#endif

// ---------- fp32 -> bf16 convert ----------
__global__ __launch_bounds__(256) void k_cvt_bf16(const float* __restrict__ in,
                                                  u16* __restrict__ out, int n) {
  int i = blockIdx.x * 256 + threadIdx.x;
  if (i < n) out[i] = f2bf(in[i]);
}

// ---------- gathered-column index table ----------
// For q-tile t: columns = covered 14-blocks (local) U all global cols (k%14>=12),
// deduped, padded with SENT. Depends only on t. 64 tiles x 192 slots.
__global__ void k_build_idx(u16* __restrict__ idx) {
  int t = threadIdx.x;
  if (t >= NN / 16) return;
  u16* row = idx + t * NSLOT;
  int lo = ((16 * t) / STRIDE_L) * STRIDE_L;
  int hi = ((16 * t + 15) / STRIDE_L + 1) * STRIDE_L;
  if (hi > NN) hi = NN;
  int cnt = 0;
  for (int k = lo; k < hi; ++k) row[cnt++] = (u16)k;
  for (int j = 0; j * STRIDE_L + STRIDE_L - 2 < NN; ++j) {
    #pragma unroll
    for (int r = STRIDE_L - 2; r < STRIDE_L; ++r) {
      int k = j * STRIDE_L + r;
      if (k < NN && !(k >= lo && k < hi)) row[cnt++] = (u16)k;
    }
  }
  while (cnt < NSLOT) row[cnt++] = (u16)SENT;
}

// ---------- shared GEMM mainloop (TDM double-buffered B panel) ----------
__device__ inline void gemm_mainloop(const u16* __restrict__ A,
                                     const u16* __restrict__ Bw,
                                     int nrows_total, int m0, int n0,
                                     u16* Bs, v8f acc[2][4]) {
  const int tid  = threadIdx.x;
  const int w    = tid >> 5;
  const int lane = tid & 31;
  const int hf   = lane >> 4;
  const int lr   = lane & 15;
  const u16* Arow0 = A + (size_t)(m0 + w * 32 + lr) * CC;
  const u16* Arow1 = Arow0 + (size_t)16 * CC;

#if HAVE_TDM
  const unsigned lds0 = (unsigned)(size_t)Bs;
  const unsigned lds1 = lds0 + 64 * CHUNK * 2;
  if (w == 0)
    tdm_load_2d(lds0, Bw + (size_t)n0 * CC, CC, (unsigned)nrows_total, CHUNK, 64, CC);
#endif

  for (int c = 0; c < NCHUNK; ++c) {
    const int cur = c & 1;
#if HAVE_TDM
    if (w == 0) {
      if (c + 1 < NCHUNK) {
        tdm_load_2d(cur ? lds0 : lds1, Bw + (size_t)n0 * CC + (c + 1) * CHUNK,
                    CC, (unsigned)nrows_total, CHUNK, 64, CC);
        __builtin_amdgcn_s_wait_tensorcnt(1);  // in-order: current buffer done
      } else {
        __builtin_amdgcn_s_wait_tensorcnt(0);
      }
    }
    __syncthreads();
#else
    {
      const int n  = tid >> 2;
      const int ko = (tid & 3) * 32;
      #pragma unroll
      for (int q = 0; q < 4; ++q)
        *(u16x8*)&Bs[(size_t)cur * 64 * CHUNK + n * CHUNK + ko + q * 8] =
            *(const u16x8*)&Bw[(size_t)(n0 + n) * CC + c * CHUNK + ko + q * 8];
      __syncthreads();
    }
#endif
    const u16* Bc = Bs + (size_t)cur * 64 * CHUNK;
    #pragma unroll
    for (int kk = 0; kk < CHUNK / 32; ++kk) {
      const int kb = c * CHUNK + kk * 32;
      v16bf a0 = load_frag(Arow0 + kb, hf);
      v16bf a1 = load_frag(Arow1 + kb, hf);
      #pragma unroll
      for (int j = 0; j < 4; ++j) {
        v16bf b = load_frag(&Bc[(j * 16 + lr) * CHUNK + kk * 32], hf);
        acc[0][j] = wmma_bf16(a0, b, acc[0][j]);
        acc[1][j] = wmma_bf16(a1, b, acc[1][j]);
      }
    }
    __syncthreads();
  }
}

// ---------- GEMM 1: qkv = x @ w_qkv^T + b -> q/k/v [B,H,N,D] bf16 ----------
__global__ __launch_bounds__(256) void k_gemm_qkv(const u16* __restrict__ A,
                                                  const u16* __restrict__ Bw,
                                                  const float* __restrict__ bias,
                                                  u16* __restrict__ qws,
                                                  u16* __restrict__ kws,
                                                  u16* __restrict__ vws) {
  __shared__ u16 Bs[2 * 64 * CHUNK];   // 32 KB
  const int n0 = blockIdx.x * 64;
  const int m0 = blockIdx.y * 256;
  v8f acc[2][4] = {};
  gemm_mainloop(A, Bw, K3, m0, n0, Bs, acc);

  const int lane = threadIdx.x & 31;
  const int w    = threadIdx.x >> 5;
  const int hf   = lane >> 4;
  const int lr   = lane & 15;
  u16* dsts[3] = { qws, kws, vws };
  #pragma unroll
  for (int t = 0; t < 2; ++t) {
    #pragma unroll
    for (int j = 0; j < 4; ++j) {
      const int n = n0 + j * 16 + lr;
      const int which = n / CC;
      const int h = (n % CC) / DD;
      const int d = n % DD;
      const float bn = bias[n];
      #pragma unroll
      for (int v = 0; v < 8; ++v) {
        const int m = m0 + w * 32 + t * 16 + v + 8 * hf;
        const int b = m / NN, row = m % NN;
        dsts[which][(((size_t)b * HH + h) * NN + row) * DD + d] = f2bf(acc[t][j][v] + bn);
      }
    }
  }
}

// ---------- GEMM 2: out = attn @ w_proj^T + b (fp32 out) ----------
__global__ __launch_bounds__(256) void k_gemm_proj(const u16* __restrict__ A,
                                                   const u16* __restrict__ Bw,
                                                   const float* __restrict__ bias,
                                                   float* __restrict__ out) {
  __shared__ u16 Bs[2 * 64 * CHUNK];   // 32 KB
  const int n0 = blockIdx.x * 64;
  const int m0 = blockIdx.y * 256;
  v8f acc[2][4] = {};
  gemm_mainloop(A, Bw, CC, m0, n0, Bs, acc);

  const int lane = threadIdx.x & 31;
  const int w    = threadIdx.x >> 5;
  const int hf   = lane >> 4;
  const int lr   = lane & 15;
  #pragma unroll
  for (int t = 0; t < 2; ++t) {
    #pragma unroll
    for (int j = 0; j < 4; ++j) {
      const int n = n0 + j * 16 + lr;
      const float bn = bias[n];
      #pragma unroll
      for (int v = 0; v < 8; ++v) {
        const int m = m0 + w * 32 + t * 16 + v + 8 * hf;
        out[(size_t)m * CC + n] = acc[t][j][v] + bn;
      }
    }
  }
}

// ---------- gathered-column flash attention ----------
// grid (8, H, B), 8 waves/block; wave w owns q-tile t = blockIdx.x*8+w.
// 6 chunks of 32 gathered key columns (vs 64 dense tiles). Per-wave private LDS,
// no block barriers. Sentinel/fully-masked chunks handled by select-exp.
__global__ __launch_bounds__(256) void k_attn(const u16* __restrict__ qws,
                                              const u16* __restrict__ kws,
                                              const u16* __restrict__ vws,
                                              const u16* __restrict__ idx,
                                              u16* __restrict__ attn_out) {
  __shared__ u16 Vt[8][DD * 32];    // per-wave gathered V^T: [d][slot]
  __shared__ u16 Ps[8][16 * 32];    // per-wave P staging: [row][slot]

  const int tid  = threadIdx.x;
  const int w    = tid >> 5;
  const int lane = tid & 31;
  const int hf   = lane >> 4;
  const int lr   = lane & 15;
  const int hd   = blockIdx.y;
  const int bb   = blockIdx.z;
  const int t    = blockIdx.x * 8 + w;
  const int q0   = t * 16;

  const size_t headoff = ((size_t)bb * HH + hd) * NN * DD;
  const u16* qp = qws + headoff;
  const u16* kp = kws + headoff;
  const u16* vp = vws + headoff;
  const u16* tix = idx + (size_t)t * NSLOT;

  const u16* Qrow = qp + (size_t)(q0 + lr) * DD;
  v16bf aq0 = load_frag(Qrow, hf);
  v16bf aq1 = load_frag(Qrow + 32, hf);

  v8f o[4] = {};
  float rmax[8], rsum[8];
  #pragma unroll
  for (int v = 0; v < 8; ++v) { rmax[v] = -1e30f; rsum[v] = 0.0f; }

  u16* myV = &Vt[w][0];
  u16* myP = &Ps[w][0];
  const float scale = 0.125f;  // 1/sqrt(64)
  const int mrow = q0 + 8 * hf;  // + v gives this lane's C-layout row

  for (int cc = 0; cc < NKC; ++cc) {
    // --- gather this wave's 32 V rows, transposed into private Vt[d][slot] ---
    {
      const u16 kr = tix[cc * 32 + lane];
      const int krow = (kr == (u16)SENT) ? 0 : (int)kr;
      const u16* Vrow = vp + (size_t)krow * DD;
      #pragma unroll
      for (int qq = 0; qq < 8; ++qq) {
        u16x8 vv = *(const u16x8*)(Vrow + qq * 8);
        #pragma unroll
        for (int i = 0; i < 8; ++i) myV[(qq * 8 + i) * 32 + lane] = vv[i];
      }
    }

    // --- S = Q @ K^T over 32 gathered keys ---
    const u16 kA16 = tix[cc * 32 + lr];
    const u16 kB16 = tix[cc * 32 + 16 + lr];
    const int rA = (kA16 == (u16)SENT) ? 0 : (int)kA16;
    const int rB = (kB16 == (u16)SENT) ? 0 : (int)kB16;
    const u16* KrowA = kp + (size_t)rA * DD;
    const u16* KrowB = kp + (size_t)rB * DD;
    v8f s0 = {}, s1 = {};
    s0 = wmma_bf16(aq0, load_frag(KrowA, hf), s0);
    s0 = wmma_bf16(aq1, load_frag(KrowA + 32, hf), s0);
    s1 = wmma_bf16(aq0, load_frag(KrowB, hf), s1);
    s1 = wmma_bf16(aq1, load_frag(KrowB + 32, hf), s1);

    // --- mask + online softmax ---
    #pragma unroll
    for (int v = 0; v < 8; ++v) {
      const int m = mrow + v;
      const bool mA = (kA16 != (u16)SENT) &&
                      ((m / STRIDE_L == rA / STRIDE_L) || (rA % STRIDE_L >= STRIDE_L - 2));
      const bool mB = (kB16 != (u16)SENT) &&
                      ((m / STRIDE_L == rB / STRIDE_L) || (rB % STRIDE_L >= STRIDE_L - 2));
      const float sv0 = mA ? s0[v] * scale : -1e30f;
      const float sv1 = mB ? s1[v] * scale : -1e30f;

      float cm = fmaxf(sv0, sv1);
      #pragma unroll
      for (int off = 1; off < 16; off <<= 1) cm = fmaxf(cm, __shfl_xor(cm, off, 32));
      const float nm   = fmaxf(rmax[v], cm);
      const float corr = __expf(rmax[v] - nm);
      rmax[v] = nm;

      const float p0 = mA ? __expf(sv0 - nm) : 0.0f;  // robust vs fully-masked chunks
      const float p1 = mB ? __expf(sv1 - nm) : 0.0f;
      float cs = p0 + p1;
      #pragma unroll
      for (int off = 1; off < 16; off <<= 1) cs += __shfl_xor(cs, off, 32);
      rsum[v] = rsum[v] * corr + cs;

      #pragma unroll
      for (int j = 0; j < 4; ++j) o[j][v] *= corr;

      myP[(v + 8 * hf) * 32 + lr]      = f2bf(p0);
      myP[(v + 8 * hf) * 32 + 16 + lr] = f2bf(p1);
    }

    // --- O += P @ V ---
    v16bf pa = load_frag(&myP[lr * 32], hf);
    #pragma unroll
    for (int j = 0; j < 4; ++j) {
      v16bf bv = load_frag(&myV[(j * 16 + lr) * 32], hf);
      o[j] = wmma_bf16(pa, bv, o[j]);
    }
  }

  #pragma unroll
  for (int v = 0; v < 8; ++v) {
    const float inv = 1.0f / rsum[v];
    const int m = mrow + v;
    #pragma unroll
    for (int j = 0; j < 4; ++j) {
      attn_out[((size_t)bb * NN + m) * CC + hd * DD + j * 16 + lr] = f2bf(o[j][v] * inv);
    }
  }
}

// ---------- launch ----------
extern "C" void kernel_launch(void* const* d_in, const int* in_sizes, int n_in,
                              void* d_out, int out_size, void* d_ws, size_t ws_size,
                              hipStream_t stream) {
  const float* x      = (const float*)d_in[0];
  const float* w_qkv  = (const float*)d_in[1];
  const float* b_qkv  = (const float*)d_in[2];
  const float* w_proj = (const float*)d_in[3];
  const float* b_proj = (const float*)d_in[4];
  float* out = (float*)d_out;

  u16* ws = (u16*)d_ws;
  const size_t nX = (size_t)MM * CC;
  u16* x_bf   = ws;
  u16* q_bf   = x_bf + nX;
  u16* k_bf   = q_bf + nX;
  u16* v_bf   = k_bf + nX;
  u16* att_bf = v_bf + nX;
  u16* wq_bf  = att_bf + nX;
  u16* wp_bf  = wq_bf + (size_t)K3 * CC;
  u16* idx_tb = wp_bf + (size_t)CC * CC;      // 64 * 192 u16

  k_cvt_bf16<<<dim3((unsigned)((nX + 255) / 256)), 256, 0, stream>>>(x, x_bf, (int)nX);
  k_cvt_bf16<<<dim3((K3 * CC + 255) / 256), 256, 0, stream>>>(w_qkv, wq_bf, K3 * CC);
  k_cvt_bf16<<<dim3((CC * CC + 255) / 256), 256, 0, stream>>>(w_proj, wp_bf, CC * CC);
  k_build_idx<<<dim3(1), 64, 0, stream>>>(idx_tb);

  k_gemm_qkv<<<dim3(K3 / 64, MM / 256), 256, 0, stream>>>(x_bf, wq_bf, b_qkv,
                                                          q_bf, k_bf, v_bf);
  k_attn<<<dim3(NN / (16 * 8), HH, BB), 256, 0, stream>>>(q_bf, k_bf, v_bf, idx_tb, att_bf);

  k_gemm_proj<<<dim3(CC / 64, MM / 256), 256, 0, stream>>>(att_bf, wp_bf, b_proj, out);
}
